// Attention_VK_27006754357511
// MI455X (gfx1250) — compile-verified, matrix operand order checked by reading, not run on hardware
//
#include <hip/hip_runtime.h>

// ---------------------------------------------------------------------------
// MI455X (gfx1250) fused prompt-KV attention
//   f16 WMMA (v_wmma_f32_16x16x32_f16) for all matmuls, fp32 accum/softmax.
//   Flash-style attention, double-buffered K/V tiles with async global->LDS
//   copies (ASYNCcnt). GEMMs use a ping-pong register pipeline (no rotation
//   moves).
// ---------------------------------------------------------------------------

typedef __attribute__((ext_vector_type(16))) _Float16 v16h;
typedef __attribute__((ext_vector_type(4)))  _Float16 v4h;
typedef __attribute__((ext_vector_type(8)))  float    v8f;

#define WMMA(a, b, c) \
  __builtin_amdgcn_wmma_f32_16x16x32_f16(false, (a), false, (b), (short)0, (c), false, false)

constexpr int B_ = 32, N_ = 785, C_ = 768, H_ = 12, D_ = 64, P_ = 50, M_ = 835;
constexpr int R_ = B_ * N_;            // 25120 rows of the token matrix
constexpr float SCALE = 0.125f;        // 1/sqrt(64), folded into q at store
constexpr float L2E   = 1.44269504088896340736f;

__device__ __forceinline__ v16h ldsv(const _Float16* p) { return *(const v16h*)p; }

// async copy 16B global -> LDS (per lane), tracked by ASYNCcnt
__device__ __forceinline__ void async_g2l_b128(const _Float16* gaddr, _Float16* lptr) {
  unsigned long long ga = (unsigned long long)(const void*)gaddr;
  unsigned loff = (unsigned)(unsigned long long)(void*)lptr;  // low 32 bits = LDS offset
  asm volatile("global_load_async_to_lds_b128 %0, %1, off"
               :: "v"(loff), "v"(ga) : "memory");
}
__device__ __forceinline__ void wait_async0() {
  asm volatile("s_wait_asynccnt 0x0" ::: "memory");
}
__device__ __forceinline__ void wait_async1() {
  asm volatile("s_wait_asynccnt 0x1" ::: "memory");
}

// --------------------------- fp32 -> f16 convert ---------------------------
__global__ void cvt_f32_f16(const float* __restrict__ s, _Float16* __restrict__ d, int n4) {
  int i = blockIdx.x * blockDim.x + threadIdx.x;
  if (i >= n4) return;
  float4 v = ((const float4*)s)[i];
  v4h o = {(_Float16)v.x, (_Float16)v.y, (_Float16)v.z, (_Float16)v.w};
  ((v4h*)d)[i] = o;
}

// ------------------- broadcast prompt tokens into K/V ----------------------
__global__ void prompt_fill(const float* __restrict__ pkv,
                            _Float16* __restrict__ kb, _Float16* __restrict__ vb) {
  int i = blockIdx.x * blockDim.x + threadIdx.x;
  const int total = B_ * H_ * P_ * D_;
  if (i >= total) return;
  int dd = i % D_; int t = i / D_;
  int p  = t % P_; t /= P_;
  int h  = t % H_; int b = t / H_;
  _Float16 val = (_Float16)pkv[(h * P_ + p) * D_ + dd];
  size_t off = (((size_t)(b * H_ + h)) * M_ + (N_ + p)) * D_ + dd;
  kb[off] = val;
  vb[off] = val;
}

// ------------------------------ QKV GEMM -----------------------------------
// Out[r, c] = sum_k xh[r,k] * wqkv[c,k]; scatter c -> (q|k|v)[b,h,n,dd].
// Block: 8 waves, each wave a 16x64 tile; ping-pong (unroll-by-2) pipeline.
__global__ __launch_bounds__(256) void qkv_gemm(const _Float16* __restrict__ xh,
                                                const _Float16* __restrict__ wh,
                                                _Float16* __restrict__ qo,
                                                _Float16* __restrict__ ko,
                                                _Float16* __restrict__ vo) {
  const int lane = threadIdx.x & 31, wave = threadIdx.x >> 5;
  const int hl = lane >> 4, l16 = lane & 15;
  const int r0 = blockIdx.y * 128 + wave * 16;
  const int c0 = blockIdx.x * 64;
  int rowA = r0 + l16; if (rowA > R_ - 1) rowA = R_ - 1;
  const _Float16* ap = xh + (size_t)rowA * C_ + hl * 16;
  const _Float16* wp = wh + (size_t)(c0 + l16) * C_ + hl * 16;

  v8f acc[4] = {};
  v16h aA = *(const v16h*)ap;
  v16h bA[4];
#pragma unroll
  for (int t = 0; t < 4; ++t) bA[t] = *(const v16h*)(wp + (size_t)t * 16 * C_);

  for (int k0 = 0; k0 < C_; k0 += 64) {           // 12 iterations
    // stage B loads (k0+32 < C_ always holds)
    v16h aB = *(const v16h*)(ap + k0 + 32);
    v16h bB[4];
#pragma unroll
    for (int t = 0; t < 4; ++t) bB[t] = *(const v16h*)(wp + (size_t)t * 16 * C_ + k0 + 32);
    __builtin_prefetch(ap + k0 + 96, 0, 3);       // global_prefetch_b8
#pragma unroll
    for (int t = 0; t < 4; ++t) acc[t] = WMMA(aA, bA[t], acc[t]);
    if (k0 + 64 < C_) {                           // next stage A loads
      aA = *(const v16h*)(ap + k0 + 64);
#pragma unroll
      for (int t = 0; t < 4; ++t) bA[t] = *(const v16h*)(wp + (size_t)t * 16 * C_ + k0 + 64);
    }
#pragma unroll
    for (int t = 0; t < 4; ++t) acc[t] = WMMA(aB, bB[t], acc[t]);
  }

#pragma unroll
  for (int t = 0; t < 4; ++t) {
    const int c = c0 + t * 16 + l16;
    const int which = c / C_;          // 0=q, 1=k, 2=v
    const int cc = c % C_;
    const int h = cc / D_, dd = cc % D_;
#pragma unroll
    for (int j = 0; j < 8; ++j) {
      const int r = r0 + j + 8 * hl;
      if (r >= R_) continue;
      const int b = r / N_, n = r % N_;
      const float val = acc[t][j];
      if (which == 0)
        qo[(((size_t)(b * H_ + h)) * N_ + n) * D_ + dd] = (_Float16)(val * SCALE);
      else if (which == 1)
        ko[(((size_t)(b * H_ + h)) * M_ + n) * D_ + dd] = (_Float16)val;
      else
        vo[(((size_t)(b * H_ + h)) * M_ + n) * D_ + dd] = (_Float16)val;
    }
  }
}

// --------------------------- flash attention -------------------------------
// grid.x = ceil(N/128) q-row blocks, grid.y = B*H. Block = 8 waves, each wave
// owns 16 q rows; m-loop tiles of 32 keys; online softmax in fp32.
// K tiles: async global->LDS (double-buffered); V tiles: load + LDS transpose
// (double-buffered). Stage i+1 copy overlaps stage i compute.
__global__ __launch_bounds__(256) void attn_kernel(const _Float16* __restrict__ q,
                                                   const _Float16* __restrict__ kb,
                                                   const _Float16* __restrict__ vb,
                                                   _Float16* __restrict__ outh) {
  __shared__ __align__(32) _Float16 lds_k[2][32 * 64];   // K tile [m][d]
  __shared__ __align__(32) _Float16 lds_vt[2][64 * 32];  // V tile transposed [d][m]
  __shared__ __align__(32) _Float16 lds_p[8][16 * 32];   // per-wave P scratch

  const int bh = blockIdx.y;
  const int b = bh / H_, h = bh % H_;
  const int lane = threadIdx.x & 31, wave = threadIdx.x >> 5;
  const int hl = lane >> 4, l16 = lane & 15;
  const int qrow0 = blockIdx.x * 128 + wave * 16;

  int rowc = qrow0 + l16; if (rowc > N_ - 1) rowc = N_ - 1;
  const _Float16* qp = q + (((size_t)bh) * N_ + rowc) * D_;
  const v16h aq0 = *(const v16h*)(qp + hl * 16);        // K = 0..31 (logical)
  const v16h aq1 = *(const v16h*)(qp + 32 + hl * 16);   // K = 32..63

  v8f acc[4] = {};
  float rmax[8], rsum[8];
#pragma unroll
  for (int j = 0; j < 8; ++j) { rmax[j] = -3.0e38f; rsum[j] = 0.f; }

  const _Float16* kbase = kb + ((size_t)bh) * M_ * D_;
  const _Float16* vbase = vb + ((size_t)bh) * M_ * D_;

  // this thread's fixed slice of the K tile: one async b128 per m-step
  const int km  = (int)threadIdx.x >> 3;          // K row within tile  [0,32)
  const int kcc = ((int)threadIdx.x & 7) * 8;     // 8-half chunk column

  const int NSTEPS = (M_ + 31) / 32;              // 27

  // stage loader: K via async copy, V via load + element transpose
  auto load_tiles = [&](int step) {
    const int m0 = step * 32;
    const int buf = step & 1;
    {
      int gm = m0 + km; if (gm > M_ - 1) gm = M_ - 1;
      async_g2l_b128(kbase + (size_t)gm * D_ + kcc, &lds_k[buf][km * 64 + kcc]);
    }
#pragma unroll
    for (int i = 0; i < 2; ++i) {
      int li = (int)threadIdx.x * 2 + i;
      int m = li >> 4;               // 16 uint2 per 64-elem row
      int col = (li & 15) << 2;
      int gm = m0 + m; if (gm > M_ - 1) gm = M_ - 1;
      union { uint2 u; _Float16 e[4]; } vd;
      vd.u = *(const uint2*)(vbase + (size_t)gm * D_ + col);
#pragma unroll
      for (int e = 0; e < 4; ++e) lds_vt[buf][(col + e) * 32 + m] = vd.e[e];
    }
  };

  load_tiles(0);

  for (int step = 0; step < NSTEPS; ++step) {
    const int m0 = step * 32;
    const int buf = step & 1;
    const bool more = (step + 1) < NSTEPS;
    if (more) load_tiles(step + 1);       // overlap next copy with this compute
    if (more) wait_async1(); else wait_async0();   // stage `step` K copy done
    __syncthreads();                      // all threads' K async + V ds-stores

    // S = Q K^T : two 16x16 n-tiles, K=64 in two WMMA steps each
    const _Float16* lk = lds_k[buf];
    v8f s0 = {}, s1 = {};
    s0 = WMMA(aq0, ldsv(&lk[l16 * 64 + hl * 16]), s0);
    s0 = WMMA(aq1, ldsv(&lk[l16 * 64 + 32 + hl * 16]), s0);
    s1 = WMMA(aq0, ldsv(&lk[(16 + l16) * 64 + hl * 16]), s1);
    s1 = WMMA(aq1, ldsv(&lk[(16 + l16) * 64 + 32 + hl * 16]), s1);

    const bool ok0 = (m0 + l16) < M_;
    const bool ok1 = (m0 + 16 + l16) < M_;

    // online softmax; D-layout row = j + 8*hl, cols spread over 16 lanes
#pragma unroll
    for (int j = 0; j < 8; ++j) {
      float a0 = ok0 ? s0[j] : -1e30f;
      float a1 = ok1 ? s1[j] : -1e30f;
      float tm = fmaxf(a0, a1);
#pragma unroll
      for (int off = 1; off < 16; off <<= 1) tm = fmaxf(tm, __shfl_xor(tm, off, 32));
      float nm = fmaxf(rmax[j], tm);
      float alpha = exp2f((rmax[j] - nm) * L2E);  // first iter: exp2(-inf)=0
      float p0 = ok0 ? exp2f((a0 - nm) * L2E) : 0.f;
      float p1 = ok1 ? exp2f((a1 - nm) * L2E) : 0.f;
      float ps = p0 + p1;
#pragma unroll
      for (int off = 1; off < 16; off <<= 1) ps += __shfl_xor(ps, off, 32);
      rsum[j] = rsum[j] * alpha + ps;
      rmax[j] = nm;
#pragma unroll
      for (int t = 0; t < 4; ++t) acc[t][j] *= alpha;
      int prow = j + 8 * hl;
      lds_p[wave][prow * 32 + l16]      = (_Float16)p0;
      lds_p[wave][prow * 32 + 16 + l16] = (_Float16)p1;
    }

    // P (A-frag via LDS transpose) x V (B-frags from transposed LDS tile)
    v16h pa = ldsv(&lds_p[wave][l16 * 32 + hl * 16]);
    const _Float16* lv = lds_vt[buf];
#pragma unroll
    for (int t = 0; t < 4; ++t)
      acc[t] = WMMA(pa, ldsv(&lv[(t * 16 + l16) * 32 + hl * 16]), acc[t]);
    __syncthreads();   // all waves done with buf before step+2 overwrites it
  }

  // epilogue: normalize, store f16 to [B, N, h*64+dd]
#pragma unroll
  for (int j = 0; j < 8; ++j) {
    float inv = 1.0f / rsum[j];
    int row = qrow0 + j + 8 * hl;
    if (row < N_) {
#pragma unroll
      for (int t = 0; t < 4; ++t) {
        int dd = t * 16 + l16;
        outh[((size_t)(b * N_ + row)) * C_ + h * D_ + dd] = (_Float16)(acc[t][j] * inv);
      }
    }
  }
}

// --------------------------- output projection -----------------------------
__global__ __launch_bounds__(256) void proj_gemm(const _Float16* __restrict__ ah,
                                                 const _Float16* __restrict__ wh,
                                                 const float* __restrict__ bias,
                                                 float* __restrict__ out) {
  const int lane = threadIdx.x & 31, wave = threadIdx.x >> 5;
  const int hl = lane >> 4, l16 = lane & 15;
  const int r0 = blockIdx.y * 128 + wave * 16;
  const int c0 = blockIdx.x * 64;
  int rowA = r0 + l16; if (rowA > R_ - 1) rowA = R_ - 1;
  const _Float16* ap = ah + (size_t)rowA * C_ + hl * 16;
  const _Float16* wp = wh + (size_t)(c0 + l16) * C_ + hl * 16;

  v8f acc[4] = {};
  v16h aA = *(const v16h*)ap;
  v16h bA[4];
#pragma unroll
  for (int t = 0; t < 4; ++t) bA[t] = *(const v16h*)(wp + (size_t)t * 16 * C_);

  for (int k0 = 0; k0 < C_; k0 += 64) {           // 12 iterations
    v16h aB = *(const v16h*)(ap + k0 + 32);
    v16h bB[4];
#pragma unroll
    for (int t = 0; t < 4; ++t) bB[t] = *(const v16h*)(wp + (size_t)t * 16 * C_ + k0 + 32);
    __builtin_prefetch(ap + k0 + 96, 0, 3);
#pragma unroll
    for (int t = 0; t < 4; ++t) acc[t] = WMMA(aA, bA[t], acc[t]);
    if (k0 + 64 < C_) {
      aA = *(const v16h*)(ap + k0 + 64);
#pragma unroll
      for (int t = 0; t < 4; ++t) bA[t] = *(const v16h*)(wp + (size_t)t * 16 * C_ + k0 + 64);
    }
#pragma unroll
    for (int t = 0; t < 4; ++t) acc[t] = WMMA(aB, bB[t], acc[t]);
  }

#pragma unroll
  for (int t = 0; t < 4; ++t) {
    const int c = c0 + t * 16 + l16;
    const float bv = bias[c];
#pragma unroll
    for (int j = 0; j < 8; ++j) {
      const int r = r0 + j + 8 * hl;
      if (r < R_) out[(size_t)r * C_ + c] = acc[t][j] + bv;
    }
  }
}

// ------------------------------- launcher ----------------------------------
extern "C" void kernel_launch(void* const* d_in, const int* in_sizes, int n_in,
                              void* d_out, int out_size, void* d_ws, size_t ws_size,
                              hipStream_t stream) {
  const float* x      = (const float*)d_in[0];
  const float* qkv_w  = (const float*)d_in[1];
  const float* proj_w = (const float*)d_in[2];
  const float* proj_b = (const float*)d_in[3];
  const float* prompt = (const float*)d_in[4];
  float* out = (float*)d_out;

  char* ws = (char*)d_ws;
  size_t off = 0;
  auto alloc = [&](size_t bytes) -> void* {
    void* p = ws + off;
    off += (bytes + 255) & ~(size_t)255;
    return p;
  };
  _Float16* xh    = (_Float16*)alloc((size_t)R_ * C_ * 2);       // 38.6 MB
  _Float16* wqkv  = (_Float16*)alloc((size_t)3 * C_ * C_ * 2);   //  3.5 MB
  _Float16* wproj = (_Float16*)alloc((size_t)C_ * C_ * 2);       //  1.2 MB
  _Float16* qb    = (_Float16*)alloc((size_t)B_ * H_ * N_ * D_ * 2);  // 38.6 MB
  _Float16* kb    = (_Float16*)alloc((size_t)B_ * H_ * M_ * D_ * 2);  // 41.0 MB
  _Float16* vb    = (_Float16*)alloc((size_t)B_ * H_ * M_ * D_ * 2);  // 41.0 MB
  _Float16* aoh   = xh;  // attn output reuses xh (x consumed by qkv_gemm)

  // 1) fp32 -> f16 converts
  {
    int n4 = R_ * C_ / 4;
    cvt_f32_f16<<<(n4 + 255) / 256, 256, 0, stream>>>(x, xh, n4);
  }
  {
    int n4 = 3 * C_ * C_ / 4;
    cvt_f32_f16<<<(n4 + 255) / 256, 256, 0, stream>>>(qkv_w, wqkv, n4);
  }
  {
    int n4 = C_ * C_ / 4;
    cvt_f32_f16<<<(n4 + 255) / 256, 256, 0, stream>>>(proj_w, wproj, n4);
  }

  // 2) QKV projection (q pre-scaled by 1/sqrt(d))
  {
    dim3 grid(3 * C_ / 64, (R_ + 127) / 128);
    qkv_gemm<<<grid, 256, 0, stream>>>(xh, wqkv, qb, kb, vb);
  }

  // 3) broadcast prompt tokens into K/V rows [N, N+P)
  {
    int n = B_ * H_ * P_ * D_;
    prompt_fill<<<(n + 255) / 256, 256, 0, stream>>>(prompt, kb, vb);
  }

  // 4) flash attention
  {
    dim3 grid((N_ + 127) / 128, B_ * H_);
    attn_kernel<<<grid, 256, 0, stream>>>(qb, kb, vb, aoh);
  }

  // 5) output projection + bias (fp32 out)
  {
    dim3 grid(C_ / 64, (R_ + 127) / 128);
    proj_gemm<<<grid, 256, 0, stream>>>(aoh, wproj, proj_b, out);
  }
}
